// GOTreeEncoder_89421219103622
// MI455X (gfx1250) — matrix operand order repaired and checked by reading, not compiled, not run on hardware
//
#include <hip/hip_runtime.h>
#include <hip/hip_bf16.h>

// ---------------------------------------------------------------------------
// GOTreeEncoder on MI455X (gfx1250, wave32)
//  - Dense GEMMs via v_wmma_f32_16x16x32_bf16, f32 accumulate
//  - A staged to zero-padded bf16 [M,Kpad]; weights pre-transposed to bf16
//    [N,Kpad]: all fragment loads are contiguous global_load_b128
//  - 2 m-tiles x 4 n-tiles per wave: 8 independent WMMA chains per K-step,
//    1.5 b128 loads per WMMA
//  - GAT edge softmax via atomic max/add passes; fused double LN + mean pool
// ---------------------------------------------------------------------------

typedef __attribute__((ext_vector_type(16))) __bf16 v16bf;
typedef __attribute__((ext_vector_type(8)))  __bf16 v8bf;
typedef __attribute__((ext_vector_type(8)))  float  v8f;

#define GEMM_MT 2   // m-tiles per wave (B-fragment reused across them)
#define GEMM_NT 4   // n-tiles per wave (A-fragment reused across them)

static inline int round_up32(int x) { return (x + 31) & ~31; }

// ---------------------------------------------------------------------------
// A-staging: dst[r, c] = bf16(src[r*Cc + c]) for c<Cc, 0 for c<Cpad (pad)
// ---------------------------------------------------------------------------
__global__ void cvt_a_bf16(const float* __restrict__ src, __bf16* __restrict__ dst,
                           int R, int Cc, int Cpad) {
  long long i = (long long)blockIdx.x * blockDim.x + threadIdx.x;
  const long long tot = (long long)R * Cpad;
  if (i >= tot) return;
  const int c = (int)(i % Cpad);
  const long long r = i / Cpad;
  dst[i] = (c < Cc) ? (__bf16)src[r * Cc + c] : (__bf16)0.0f;
}

// ---------------------------------------------------------------------------
// Weight transpose: Bt[n, k] = bf16(B[k*Nn + n]) for k<K, 0 for k<Kpad
// ---------------------------------------------------------------------------
__global__ void cvt_bt_bf16(const float* __restrict__ B, __bf16* __restrict__ Bt,
                            int K, int Nn, int Kpad) {
  long long i = (long long)blockIdx.x * blockDim.x + threadIdx.x;
  const long long tot = (long long)Nn * Kpad;
  if (i >= tot) return;
  const int k = (int)(i % Kpad);
  const int n = (int)(i / Kpad);
  Bt[i] = (k < K) ? (__bf16)B[(size_t)k * Nn + n] : (__bf16)0.0f;
}

// ---------------------------------------------------------------------------
// WMMA GEMM: C[M,N] = A[M,Kp] * Bt[N,Kp]^T (+ bias[n]).  A,Bt bf16, Kp%32==0.
// One wave per block; wave computes (16*GEMM_MT) x (16*GEMM_NT) outputs.
// ---------------------------------------------------------------------------
__global__ __launch_bounds__(32)
void wmma_gemm_bf16(const __bf16* __restrict__ A, const __bf16* __restrict__ Bt,
                    float* __restrict__ C, const float* __restrict__ bias,
                    int M, int Kp, int N, int ldc) {
  const int lane = threadIdx.x & 31;
  const int hs   = lane >> 4;      // lane half: 0 => lanes 0-15, 1 => 16-31
  const int l    = lane & 15;
  const int m0   = blockIdx.x * (16 * GEMM_MT);
  const int n0   = blockIdx.y * (16 * GEMM_NT);

  const __bf16* __restrict__ Ar[GEMM_MT];
#pragma unroll
  for (int mi = 0; mi < GEMM_MT; ++mi) {
    int arow = m0 + mi * 16 + l;
    if (arow >= M) arow = M - 1;               // clamp: loads stay in-bounds
    Ar[mi] = A + (size_t)arow * Kp;
  }

  const __bf16* __restrict__ Bc[GEMM_NT];
#pragma unroll
  for (int t = 0; t < GEMM_NT; ++t)
    Bc[t] = Bt + (size_t)(n0 + t * 16 + l) * Kp;

  v8f acc[GEMM_MT][GEMM_NT];
#pragma unroll
  for (int mi = 0; mi < GEMM_MT; ++mi)
#pragma unroll
    for (int t = 0; t < GEMM_NT; ++t) acc[mi][t] = v8f{};

  for (int kb = 0; kb < Kp; kb += 32) {
    // A 16x32 bf16 fragments: contiguous K = hs*16..+15 (matches B)
    v16bf afrag[GEMM_MT];
#pragma unroll
    for (int mi = 0; mi < GEMM_MT; ++mi) {
      afrag[mi] = *(const v16bf*)(Ar[mi] + kb + hs * 16);
    }
#pragma unroll
    for (int t = 0; t < GEMM_NT; ++t) {
      // B 32x16 fragment: lane = column, contiguous K = hs*16..+15
      const v16bf bfrag = *(const v16bf*)(Bc[t] + kb + hs * 16);
#pragma unroll
      for (int mi = 0; mi < GEMM_MT; ++mi) {
        acc[mi][t] = __builtin_amdgcn_wmma_f32_16x16x32_bf16(
            false, afrag[mi], false, bfrag, (short)0, acc[mi][t], false, false);
      }
    }
  }

  // D store: lane = column, VGPR r = row (hs*8 + r)
#pragma unroll
  for (int mi = 0; mi < GEMM_MT; ++mi) {
#pragma unroll
    for (int t = 0; t < GEMM_NT; ++t) {
      const int col = n0 + t * 16 + l;
      const float bb = bias ? bias[col] : 0.0f;
#pragma unroll
      for (int r = 0; r < 8; ++r) {
        const int rr = m0 + mi * 16 + hs * 8 + r;
        if (rr < M) C[(size_t)rr * ldc + col] = acc[mi][t][r] + bb;
      }
    }
  }
}

// ---------------------------------------------------------------------------
// Helpers / elementwise kernels
// ---------------------------------------------------------------------------
__global__ void fill_f32(float* p, float v, long long n) {
  long long i = (long long)blockIdx.x * blockDim.x + threadIdx.x;
  const long long stride = (long long)gridDim.x * blockDim.x;
  for (; i < n; i += stride) p[i] = v;
}

__global__ void attn_scores(const float* __restrict__ xl,
                            const float* __restrict__ att_src,
                            const float* __restrict__ att_dst,
                            float* __restrict__ a_src, float* __restrict__ a_dst,
                            int N, int H, int C) {
  const int i = blockIdx.x * blockDim.x + threadIdx.x;
  if (i >= N * H) return;
  const int n = i / H, h = i % H;
  const float* x  = xl + ((size_t)n * H + h) * C;
  const float* ws = att_src + (size_t)h * C;
  const float* wd = att_dst + (size_t)h * C;
  float s = 0.f, d = 0.f;
  for (int c = 0; c < C; ++c) { const float v = x[c]; s += v * ws[c]; d += v * wd[c]; }
  a_src[i] = s; a_dst[i] = d;
}

__device__ __forceinline__ void atomicMaxF32(float* addr, float val) {
  int old = __float_as_int(*addr);
  while (__int_as_float(old) < val) {
    const int prev = atomicCAS((int*)addr, old, __float_as_int(val));
    if (prev == old) break;
    old = prev;
  }
}

__device__ __forceinline__ void edge_sd(const int* ei, int E, int t, int& s, int& d) {
  if (t < E) { s = ei[t]; d = ei[E + t]; }
  else       { s = t - E; d = t - E; }      // implicit self-loops
}

__global__ void edge_logits(const int* __restrict__ ei, int E, int Etot,
                            const float* __restrict__ a_src,
                            const float* __restrict__ a_dst,
                            float* __restrict__ e, float* __restrict__ emax, int H) {
  const int i = blockIdx.x * blockDim.x + threadIdx.x;
  if (i >= Etot * H) return;
  const int t = i / H, h = i % H;
  int s, d; edge_sd(ei, E, t, s, d);
  float v = a_src[s * H + h] + a_dst[d * H + h];
  v = (v > 0.f) ? v : 0.2f * v;
  e[i] = v;
  atomicMaxF32(&emax[d * H + h], v);
}

__global__ void edge_exp(const int* __restrict__ ei, int E, int Etot,
                         float* __restrict__ e, const float* __restrict__ emax,
                         float* __restrict__ denom, int H) {
  const int i = blockIdx.x * blockDim.x + threadIdx.x;
  if (i >= Etot * H) return;
  const int t = i / H, h = i % H;
  int s, d; edge_sd(ei, E, t, s, d);
  float m = emax[d * H + h];
  if (m < -1e30f) m = 0.f;                    // where(isfinite(emax), emax, 0)
  const float ex = __expf(e[i] - m);
  e[i] = ex;
  atomicAdd(&denom[d * H + h], ex);
}

__global__ void edge_scatter(const int* __restrict__ ei, int E, int Etot,
                             const float* __restrict__ e,
                             const float* __restrict__ denom,
                             const float* __restrict__ xl,
                             float* __restrict__ out, int H, int C) {
  const int cg = C >> 2;
  const long long tot = (long long)Etot * H * cg;
  long long i = (long long)blockIdx.x * blockDim.x + threadIdx.x;
  if (i >= tot) return;
  const int c4 = (int)(i % cg);
  const int h  = (int)((i / cg) % H);
  const int t  = (int)(i / ((long long)cg * H));
  int s, d; edge_sd(ei, E, t, s, d);
  const float alpha = e[(size_t)t * H + h] / (denom[d * H + h] + 1e-16f);
  const float* xs = xl  + ((size_t)s * H + h) * C + c4 * 4;
  float*       od = out + ((size_t)d * H + h) * C + c4 * 4;
  atomicAdd(&od[0], xs[0] * alpha);
  atomicAdd(&od[1], xs[1] * alpha);
  atomicAdd(&od[2], xs[2] * alpha);
  atomicAdd(&od[3], xs[3] * alpha);
}

__global__ void head_mean_bias(const float* __restrict__ acc, float* __restrict__ out,
                               const float* __restrict__ bias,
                               int N, int H, int C, int ldo) {
  const int i = blockIdx.x * blockDim.x + threadIdx.x;
  if (i >= N * C) return;
  const int n = i / C, c = i % C;
  float s = 0.f;
  for (int h = 0; h < H; ++h) s += acc[((size_t)n * H + h) * C + c];
  out[(size_t)n * ldo + c] = s / (float)H + bias[c];
}

__global__ void bias_relu_inplace(float* __restrict__ x, const float* __restrict__ bias,
                                  int N, int HC) {
  const long long i = (long long)blockIdx.x * blockDim.x + threadIdx.x;
  if (i >= (long long)N * HC) return;
  const int j = (int)(i % HC);
  const float v = x[i] + bias[j];
  x[i] = v > 0.f ? v : 0.f;
}

__global__ void double_layernorm(float* __restrict__ x,
                                 const float* __restrict__ g1, const float* __restrict__ b1,
                                 const float* __restrict__ g2, const float* __restrict__ b2,
                                 int C) {
  __shared__ float sm[256];
  const int n = blockIdx.x, c = threadIdx.x;
  float v = x[(size_t)n * C + c];
#pragma unroll
  for (int pass = 0; pass < 2; ++pass) {
    const float* g = pass ? g2 : g1;
    const float* b = pass ? b2 : b1;
    sm[c] = v; __syncthreads();
    for (int o = 128; o > 0; o >>= 1) { if (c < o) sm[c] += sm[c + o]; __syncthreads(); }
    const float mu = sm[0] / (float)C; __syncthreads();
    const float dv = v - mu;
    sm[c] = dv * dv; __syncthreads();
    for (int o = 128; o > 0; o >>= 1) { if (c < o) sm[c] += sm[c + o]; __syncthreads(); }
    const float var = sm[0] / (float)C; __syncthreads();
    v = dv * rsqrtf(var + 1e-5f) * g[c] + b[c];
  }
  x[(size_t)n * C + c] = v;
}

__global__ void mean_pool(const float* __restrict__ x, float* __restrict__ out,
                          int N, int C) {
  const long long i = (long long)blockIdx.x * blockDim.x + threadIdx.x;
  if (i >= (long long)N * C) return;
  const int c = (int)(i % C);
  atomicAdd(&out[c], x[i] / (float)N);
}

// ---------------------------------------------------------------------------
extern "C" void kernel_launch(void* const* d_in, const int* in_sizes, int n_in,
                              void* d_out, int out_size, void* d_ws, size_t ws_size,
                              hipStream_t stream) {
  const float* multi_hot = (const float*)d_in[0];
  const float* bert_feat = (const float*)d_in[1];
  const int*   ei        = (const int*)  d_in[2];
  const float* W_struct  = (const float*)d_in[3];
  const float* b_struct  = (const float*)d_in[4];
  const float* W_bert    = (const float*)d_in[5];
  const float* b_bert    = (const float*)d_in[6];
  const float* W_g1      = (const float*)d_in[7];
  const float* att_src1  = (const float*)d_in[8];
  const float* att_dst1  = (const float*)d_in[9];
  const float* b_g1      = (const float*)d_in[10];
  const float* W_g2      = (const float*)d_in[11];
  const float* att_src2  = (const float*)d_in[12];
  const float* att_dst2  = (const float*)d_in[13];
  const float* b_g2      = (const float*)d_in[14];
  const float* W_g3      = (const float*)d_in[15];
  const float* att_src3  = (const float*)d_in[16];
  const float* att_dst3  = (const float*)d_in[17];
  const float* b_g3      = (const float*)d_in[18];
  const float* ln1_g     = (const float*)d_in[19];
  const float* ln1_b     = (const float*)d_in[20];
  const float* ln2_g     = (const float*)d_in[21];
  const float* ln2_b     = (const float*)d_in[22];

  const int N  = in_sizes[1] / 768;       // 50000
  const int AW = in_sizes[0] / N;         // 2000
  const int E  = in_sizes[2] / 2;         // 200000
  const int Etot = E + N;                 // + self loops
  const int AWp = round_up32(AW);         // 2016

  // ---- workspace layout ----
  char* w = (char*)d_ws;
  const size_t big = (size_t)N * 1024;
  float*  bufA  = (float*)w;                    w += big * sizeof(float);
  float*  bufB  = (float*)w;                    w += big * sizeof(float);
  float*  asrc  = (float*)w;                    w += (size_t)N * 4 * sizeof(float);
  float*  adst  = (float*)w;                    w += (size_t)N * 4 * sizeof(float);
  float*  emax  = (float*)w;                    w += (size_t)N * 4 * sizeof(float);
  float*  denom = (float*)w;                    w += (size_t)N * 4 * sizeof(float);
  float*  ebuf  = (float*)w;                    w += (size_t)Etot * 4 * sizeof(float);
  __bf16* Xbf   = (__bf16*)w;                   w += (size_t)N * (AWp > 1024 ? AWp : 1024) * sizeof(__bf16);
  __bf16* Wbf   = (__bf16*)w;                   // 1024*1024 bf16

  auto cvtA = [&](const float* src, int R, int Cc, int Cpad) {
    const long long tot = (long long)R * Cpad;
    cvt_a_bf16<<<(int)((tot + 255) / 256), 256, 0, stream>>>(src, Xbf, R, Cc, Cpad);
  };
  auto cvtB = [&](const float* B, int K, int Nn, int Kpad) {
    const long long tot = (long long)Nn * Kpad;
    cvt_bt_bf16<<<(int)((tot + 255) / 256), 256, 0, stream>>>(B, Wbf, K, Nn, Kpad);
  };
  auto gemm = [&](float* C, const float* bias, int M, int Kp, int Nn, int ldc) {
    dim3 grid((M + 16 * GEMM_MT - 1) / (16 * GEMM_MT), (Nn / 16) / GEMM_NT);
    wmma_gemm_bf16<<<grid, 32, 0, stream>>>(Xbf, Wbf, C, bias, M, Kp, Nn, ldc);
  };
  auto fill = [&](float* p, float v, long long n) {
    long long blk = (n + 255) / 256; if (blk > 65535) blk = 65535;
    fill_f32<<<(int)blk, 256, 0, stream>>>(p, v, n);
  };
  auto gat_edges = [&](const float* xl, const float* aS, const float* aD,
                       float* acc, int H, int C) {
    attn_scores<<<(N * H + 255) / 256, 256, 0, stream>>>(xl, aS, aD, asrc, adst, N, H, C);
    fill(emax, -3.0e38f, (long long)N * H);
    fill(denom, 0.0f,    (long long)N * H);
    fill(acc,  0.0f,     (long long)N * H * C);
    const int eh = Etot * H;
    edge_logits<<<(eh + 255) / 256, 256, 0, stream>>>(ei, E, Etot, asrc, adst, ebuf, emax, H);
    edge_exp   <<<(eh + 255) / 256, 256, 0, stream>>>(ei, E, Etot, ebuf, emax, denom, H);
    const long long tot = (long long)Etot * H * (C / 4);
    edge_scatter<<<(int)((tot + 255) / 256), 256, 0, stream>>>(ei, E, Etot, ebuf, denom,
                                                               xl, acc, H, C);
  };

  // ---- fused = [struct_proj | bert_proj] -> bufA [N,256] ----
  cvtA(multi_hot, N, AW, AWp);  cvtB(W_struct, AW, 128, AWp);
  gemm(bufA,       b_struct, N, AWp, 128, 256);
  cvtA(bert_feat, N, 768, 768); cvtB(W_bert, 768, 128, 768);
  gemm(bufA + 128, b_bert,   N, 768, 128, 256);

  // ---- GAT1: 256 -> (H=2,C=128), concat=False -> x1 = bufB [N,128] ----
  cvtA(bufA, N, 256, 256);      cvtB(W_g1, 256, 256, 256);
  gemm(bufB, nullptr, N, 256, 256, 256);                        // xl1 in bufB
  gat_edges(bufB, att_src1, att_dst1, bufA, 2, 128);            // acc in bufA
  head_mean_bias<<<(N * 128 + 255) / 256, 256, 0, stream>>>(bufA, bufB, b_g1, N, 2, 128, 128);

  // ---- GAT2a: 128 -> (H=4,C=256), concat=True + ReLU -> x2 = bufB [N,1024] ----
  cvtA(bufB, N, 128, 128);      cvtB(W_g2, 128, 1024, 128);
  gemm(bufA, nullptr, N, 128, 1024, 1024);                      // xl2 in bufA
  gat_edges(bufA, att_src2, att_dst2, bufB, 4, 256);            // acc in bufB
  bias_relu_inplace<<<(int)(((long long)N * 1024 + 255) / 256), 256, 0, stream>>>(bufB, b_g2, N, 1024);

  // ---- GAT2b: 1024 -> (H=4,C=256), concat=False -> x3 = bufA [N,256] ----
  cvtA(bufB, N, 1024, 1024);    cvtB(W_g3, 1024, 1024, 1024);
  gemm(bufA, nullptr, N, 1024, 1024, 1024);                     // xl3 in bufA
  gat_edges(bufA, att_src3, att_dst3, bufB, 4, 256);            // acc in bufB
  head_mean_bias<<<(N * 256 + 255) / 256, 256, 0, stream>>>(bufB, bufA, b_g3, N, 4, 256, 256);

  // ---- LayerNorm x2 (fused) + global mean pool ----
  double_layernorm<<<N, 256, 0, stream>>>(bufA, ln1_g, ln1_b, ln2_g, ln2_b, 256);
  fill((float*)d_out, 0.0f, 256);
  mean_pool<<<(int)(((long long)N * 256 + 255) / 256), 256, 0, stream>>>(bufA, (float*)d_out, N, 256);
}
